// StratifiedAttention_7378753815106
// MI455X (gfx1250) — matrix operand order repaired and checked by reading, not compiled, not run on hardware
//
#include <hip/hip_runtime.h>
#include <hip/hip_bf16.h>

typedef __bf16 bf16;
typedef __bf16 v16bf __attribute__((ext_vector_type(16)));
typedef float  v8f   __attribute__((ext_vector_type(8)));
typedef unsigned int u32x4 __attribute__((ext_vector_type(4)));
typedef float f32x4 __attribute__((ext_vector_type(4)));

#define H      1024
#define NH     16
#define HD     64
#define BATCH  2
#define SEQ    2048
#define MROWS  (BATCH*SEQ)   /* 4096 */

/* CDNA5 async global->LDS path (ASYNCcnt), with synchronous fallback */
#if defined(__gfx1250__) && __has_builtin(__builtin_amdgcn_global_load_async_to_lds_b128) && __has_builtin(__builtin_amdgcn_s_wait_asynccnt)
#define USE_ASYNC_LDS 1
#else
#define USE_ASYNC_LDS 0
#endif

#if USE_ASYNC_LDS
typedef int v4i_vs __attribute__((vector_size(16)));
typedef __attribute__((address_space(1))) v4i_vs* gvp_t;   /* global src  */
typedef __attribute__((address_space(3))) v4i_vs* lvp_t;   /* LDS dest    */
#endif

__device__ __forceinline__ void cp16(bf16* dst_lds, const bf16* src_glb) {
#if USE_ASYNC_LDS
    __builtin_amdgcn_global_load_async_to_lds_b128(
        (gvp_t)(void*)src_glb, (lvp_t)(void*)dst_lds, 0, 0);
#else
    *(u32x4*)dst_lds = *(const u32x4*)src_glb;
#endif
}

__device__ __forceinline__ void async_wait_tile() {
#if USE_ASYNC_LDS
    __builtin_amdgcn_s_wait_asynccnt(8);   /* next tile's 8 issues may remain */
#endif
}
__device__ __forceinline__ void async_wait_all() {
#if USE_ASYNC_LDS
    __builtin_amdgcn_s_wait_asynccnt(0);
#endif
}

__device__ __forceinline__ v8f vzero8() {
    v8f v;
#pragma unroll
    for (int i = 0; i < 8; ++i) v[i] = 0.0f;
    return v;
}

/* ------------------------------------------------------------------ */
/* elementwise prep kernels                                            */
/* ------------------------------------------------------------------ */
__global__ void k_cvt_f32_bf16(const float* __restrict__ s, bf16* __restrict__ d, int n) {
    int i = (blockIdx.x * blockDim.x + threadIdx.x) * 4;
    if (i + 3 < n) {
        f32x4 f = *(const f32x4*)(s + i);
        d[i + 0] = (bf16)f.x; d[i + 1] = (bf16)f.y;
        d[i + 2] = (bf16)f.z; d[i + 3] = (bf16)f.w;
    }
}

__global__ void k_avg3_bf16(const float* __restrict__ s, bf16* __restrict__ d, int stride, int n) {
    int i = blockIdx.x * blockDim.x + threadIdx.x;
    if (i < n) d[i] = (bf16)((s[i] + s[i + stride] + s[i + 2 * stride]) * (1.0f / 3.0f));
}

__global__ void k_avg3_f32(const float* __restrict__ s, float* __restrict__ d, int stride, int n) {
    int i = blockIdx.x * blockDim.x + threadIdx.x;
    if (i < n) d[i] = (s[i] + s[i + stride] + s[i + 2 * stride]) * (1.0f / 3.0f);
}

/* ------------------------------------------------------------------ */
/* bf16 WMMA GEMM:  D[M,N] = A[M,K] @ B[N,K]^T + bias[N]               */
/* 128x128x64 tiles, 256 threads = 8 wave32s, wave tile 32x64.         */
/* Double-buffered LDS (64 KB) fed by async global->LDS copies.        */
/* XOR swizzle on 16B units: unit ^= (row & 7)  (no padding needed).   */
/* ------------------------------------------------------------------ */
#define BM 128
#define BN 128
#define BK 64

__device__ __forceinline__ void stage_rows(const bf16* __restrict__ g, bf16* __restrict__ l,
                                           int K, int row0, int kt, int lr, int uc) {
#pragma unroll
    for (int p = 0; p < 4; ++p) {
        int r = lr + p * 32;
        cp16(l + r * BK + (((uc ^ (r & 7))) << 3),
             g + (size_t)(row0 + r) * K + kt + uc * 8);
    }
}

template<int OUT_F32>
__global__ __launch_bounds__(256)
void gemm_bias_wmma(const bf16* __restrict__ A, const bf16* __restrict__ B,
                    const float* __restrict__ bias, void* __restrict__ D,
                    int M, int N, int K)
{
    __shared__ bf16 As[2][BM * BK];
    __shared__ bf16 Bs[2][BN * BK];

    const int tid  = threadIdx.x;
    const int lane = tid & 31;
    const int wid  = tid >> 5;
    const int wm   = wid & 3;      /* 4 row-slices of 32 */
    const int wn   = wid >> 2;     /* 2 col-slices of 64 */
    const int m0   = blockIdx.y * BM;
    const int n0   = blockIdx.x * BN;
    const int hw   = lane >> 4;
    const int l16  = lane & 15;

    const int lr = tid >> 3;       /* 0..31 : staging row  */
    const int uc = tid & 7;        /* 16B unit within row  */

    v8f acc[2][4];
#pragma unroll
    for (int i = 0; i < 2; ++i)
#pragma unroll
        for (int j = 0; j < 4; ++j) acc[i][j] = vzero8();

    /* prologue: stage tile 0 */
    stage_rows(A, As[0], K, m0, 0, lr, uc);
    stage_rows(B, Bs[0], K, n0, 0, lr, uc);

    const int NT = K / BK;
    for (int it = 0; it < NT; ++it) {
        const int cur = it & 1;
        if (it + 1 < NT) {   /* issue next tile, then wait only for current */
            stage_rows(A, As[cur ^ 1], K, m0, (it + 1) * BK, lr, uc);
            stage_rows(B, Bs[cur ^ 1], K, n0, (it + 1) * BK, lr, uc);
            async_wait_tile();
        } else {
            async_wait_all();
        }
        __syncthreads();

        const bf16* Ac = &As[cur][0];
        const bf16* Bc = &Bs[cur][0];
#pragma unroll
        for (int kk = 0; kk < BK; kk += 32) {
            union { v16bf v; u32x4 q[2]; } af[2];
#pragma unroll
            for (int i = 0; i < 2; ++i) {
                int row = wm * 32 + i * 16 + l16;
                int rx  = row & 7;
                int ua  = (kk >> 3) + hw;
                af[i].q[0] = *(const u32x4*)(Ac + row * BK + (((ua    ) ^ rx) << 3));
                af[i].q[1] = *(const u32x4*)(Ac + row * BK + (((ua + 2) ^ rx) << 3));
            }
            union { v16bf v; u32x4 q[2]; } bfg[4];
#pragma unroll
            for (int j = 0; j < 4; ++j) {
                int row = wn * 64 + j * 16 + l16;
                int rx  = row & 7;
                int ub  = (kk >> 3) + hw;
                bfg[j].q[0] = *(const u32x4*)(Bc + row * BK + (((ub    ) ^ rx) << 3));
                bfg[j].q[1] = *(const u32x4*)(Bc + row * BK + (((ub + 2) ^ rx) << 3));
            }
#pragma unroll
            for (int i = 0; i < 2; ++i)
#pragma unroll
                for (int j = 0; j < 4; ++j)
                    acc[i][j] = __builtin_amdgcn_wmma_f32_16x16x32_bf16(
                        false, af[i].v, false, bfg[j].v, (short)0, acc[i][j], false, false);
        }
        __syncthreads();
    }

    /* epilogue: C/D layout row = v + 8*(lane>>4), col = lane&15 */
#pragma unroll
    for (int i = 0; i < 2; ++i) {
        int mrow0 = m0 + wm * 32 + i * 16 + hw * 8;
#pragma unroll
        for (int j = 0; j < 4; ++j) {
            int col  = n0 + wn * 64 + j * 16 + l16;
            float bc = bias[col];
#pragma unroll
            for (int v = 0; v < 8; ++v) {
                float  val = acc[i][j][v] + bc;
                size_t idx = (size_t)(mrow0 + v) * N + col;
                if (OUT_F32) ((float*)D)[idx] = val;
                else         ((bf16*)D)[idx]  = (bf16)val;
            }
        }
    }
}

/* ------------------------------------------------------------------ */
/* Flash attention over fused QKV buffer [MROWS, 3H] (bf16)            */
/* grid: (SEQ/64, NH, BATCH), block 128 = 4 waves; each wave owns 16 q */
/* ------------------------------------------------------------------ */
#define FB 64
#define FLD 72

__global__ __launch_bounds__(128)
void flash_attn(const bf16* __restrict__ qkv, bf16* __restrict__ attn)
{
    __shared__ bf16 Ks[FB * FLD];        /* [key][hd]             */
    __shared__ bf16 Vs[HD * FLD];        /* transposed: [hd][key] */
    __shared__ bf16 Ps[4 * 16 * FLD];    /* per-wave P staging    */

    const int tid  = threadIdx.x;
    const int lane = tid & 31;
    const int wave = tid >> 5;
    const int hw   = lane >> 4;
    const int l16  = lane & 15;

    const int qbase = blockIdx.x * FB;
    const int h     = blockIdx.y;
    const int b     = blockIdx.z;

    const size_t rs = (size_t)3 * H;     /* QKV row stride (elements) */
    const bf16* Qg = qkv + (size_t)(b * SEQ + qbase) * rs + h * HD;
    const bf16* Kg = qkv + (size_t)(b * SEQ) * rs + H     + h * HD;
    const bf16* Vg = qkv + (size_t)(b * SEQ) * rs + 2 * H + h * HD;

    /* preload this wave's 16 q rows as two K=32 A-fragments */
    union { v16bf v; u32x4 q[2]; } aq[2];
#pragma unroll
    for (int kki = 0; kki < 2; ++kki) {
        const bf16* p = Qg + (size_t)(wave * 16 + l16) * rs + kki * 32 + hw * 8;
        aq[kki].q[0] = *(const u32x4*)p;
        aq[kki].q[1] = *(const u32x4*)(p + 16);
    }

    float m_i[8], l_i[8];
    v8f o[4];
#pragma unroll
    for (int v = 0; v < 8; ++v) { m_i[v] = -3.0e38f; l_i[v] = 0.0f; }
#pragma unroll
    for (int j = 0; j < 4; ++j) o[j] = vzero8();

    const int lr = tid >> 3;        /* 0..15 */
    const int lc = (tid & 7) * 8;

    for (int kt = 0; kt < SEQ; kt += FB) {
        /* async-stage K tile; DMA runs while we transpose V below */
#pragma unroll
        for (int p = 0; p < 4; ++p) {
            int r = lr + p * 16;
            cp16(&Ks[r * FLD + lc], Kg + (size_t)(kt + r) * rs + lc);
        }
        /* stage V transposed (needs VALU shuffle, stays synchronous) */
#pragma unroll
        for (int p = 0; p < 4; ++p) {
            int r = lr + p * 16;
            union { u32x4 q; bf16 e[8]; } vv;
            vv.q = *(const u32x4*)(Vg + (size_t)(kt + r) * rs + lc);
#pragma unroll
            for (int e = 0; e < 8; ++e) Vs[(lc + e) * FLD + r] = vv.e[e];
        }
        async_wait_all();
        __syncthreads();

        /* S = Q @ K^T  (16 x 64 per wave, 4 accumulators) */
        v8f sacc[4];
#pragma unroll
        for (int j = 0; j < 4; ++j) sacc[j] = vzero8();
#pragma unroll
        for (int kki = 0; kki < 2; ++kki)
#pragma unroll
            for (int j = 0; j < 4; ++j) {
                union { v16bf v; u32x4 q[2]; } bk;
                const bf16* p = &Ks[(j * 16 + l16) * FLD + kki * 32 + hw * 8];
                bk.q[0] = *(const u32x4*)p;
                bk.q[1] = *(const u32x4*)(p + 16);
                sacc[j] = __builtin_amdgcn_wmma_f32_16x16x32_bf16(
                    false, aq[kki].v, false, bk.v, (short)0, sacc[j], false, false);
            }

        /* online softmax: per-row stats via half-wave shuffle reductions */
        float s[4][8];
#pragma unroll
        for (int v = 0; v < 8; ++v) {
            float loc = -3.0e38f;
#pragma unroll
            for (int j = 0; j < 4; ++j) {
                float x = sacc[j][v] * 0.125f;   /* 1/sqrt(HD) */
                s[j][v] = x;
                loc = fmaxf(loc, x);
            }
            for (int off = 8; off >= 1; off >>= 1)
                loc = fmaxf(loc, __shfl_xor(loc, off, 32));
            float mn   = fmaxf(m_i[v], loc);
            float corr = __expf(m_i[v] - mn);
            m_i[v] = mn;
            float rsum = 0.0f;
#pragma unroll
            for (int j = 0; j < 4; ++j) {
                float pj = __expf(s[j][v] - mn);
                s[j][v] = pj;
                rsum += pj;
            }
            for (int off = 8; off >= 1; off >>= 1)
                rsum += __shfl_xor(rsum, off, 32);
            l_i[v] = l_i[v] * corr + rsum;
#pragma unroll
            for (int jo = 0; jo < 4; ++jo) o[jo][v] *= corr;
        }

        /* restage P (C-layout -> A-layout) through wave-private LDS */
        bf16* Pw = &Ps[wave * 16 * FLD];
#pragma unroll
        for (int j = 0; j < 4; ++j)
#pragma unroll
            for (int v = 0; v < 8; ++v)
                Pw[(v + 8 * hw) * FLD + j * 16 + l16] = (bf16)s[j][v];
        asm volatile("" ::: "memory");   /* keep LDS write/read order */

        /* O += P @ V */
#pragma unroll
        for (int kki = 0; kki < 2; ++kki) {
            union { v16bf v; u32x4 q[2]; } ap;
            const bf16* pp = &Pw[l16 * FLD + kki * 32 + hw * 8];
            ap.q[0] = *(const u32x4*)pp;
            ap.q[1] = *(const u32x4*)(pp + 16);
#pragma unroll
            for (int jo = 0; jo < 4; ++jo) {
                union { v16bf v; u32x4 q[2]; } bv;
                const bf16* pv = &Vs[(jo * 16 + l16) * FLD + kki * 32 + hw * 8];
                bv.q[0] = *(const u32x4*)pv;
                bv.q[1] = *(const u32x4*)(pv + 16);
                o[jo] = __builtin_amdgcn_wmma_f32_16x16x32_bf16(
                    false, ap.v, false, bv.v, (short)0, o[jo], false, false);
            }
        }
        __syncthreads();
    }

    /* write attn_output[b, q, h*HD + d] as bf16 */
#pragma unroll
    for (int jo = 0; jo < 4; ++jo)
#pragma unroll
        for (int v = 0; v < 8; ++v) {
            float val = o[jo][v] / l_i[v];
            attn[(size_t)(b * SEQ + qbase + wave * 16 + v + 8 * hw) * H
                 + h * HD + jo * 16 + l16] = (bf16)val;
        }
}

/* ------------------------------------------------------------------ */
extern "C" void kernel_launch(void* const* d_in, const int* in_sizes, int n_in,
                              void* d_out, int out_size, void* d_ws, size_t ws_size,
                              hipStream_t stream)
{
    (void)in_sizes; (void)n_in; (void)out_size; (void)ws_size;

    const float* hs = (const float*)d_in[0];
    const float* Wq = (const float*)d_in[1];
    const float* bq = (const float*)d_in[2];
    const float* Wk = (const float*)d_in[3];
    const float* bk = (const float*)d_in[4];
    const float* Wv = (const float*)d_in[5];
    const float* bv = (const float*)d_in[6];
    const float* Ws = (const float*)d_in[7];
    const float* bs = (const float*)d_in[8];
    const float* Wo = (const float*)d_in[9];
    const float* bo = (const float*)d_in[10];

    char*  ws  = (char*)d_ws;
    size_t off = 0;
    auto alloc = [&](size_t bytes) -> void* {
        void* p = ws + off;
        off += (bytes + 255) & ~(size_t)255;
        return p;
    };
    const int HH = H * H;
    bf16*  Xbf  = (bf16*) alloc((size_t)MROWS * H * 2);
    bf16*  Wqkv = (bf16*) alloc((size_t)3 * HH * 2);
    float* bqkv = (float*)alloc((size_t)3 * H * 4);
    bf16*  Wavg = (bf16*) alloc((size_t)HH * 2);
    float* bavg = (float*)alloc((size_t)H * 4);
    bf16*  Wob  = (bf16*) alloc((size_t)HH * 2);
    bf16*  QKV  = (bf16*) alloc((size_t)MROWS * 3 * H * 2);
    bf16*  Attn = (bf16*) alloc((size_t)MROWS * H * 2);
    bf16*  Enh  = (bf16*) alloc((size_t)MROWS * H * 2);

    const int TPB = 256;
    auto g4 = [](int n) { return (n / 4 + TPB - 1) / TPB; };

    /* precision downconvert + strata folding */
    k_cvt_f32_bf16<<<g4(MROWS * H), TPB, 0, stream>>>(hs, Xbf, MROWS * H);
    k_cvt_f32_bf16<<<g4(HH), TPB, 0, stream>>>(Wq, Wqkv,          HH);
    k_cvt_f32_bf16<<<g4(HH), TPB, 0, stream>>>(Wk, Wqkv + HH,     HH);
    k_cvt_f32_bf16<<<g4(HH), TPB, 0, stream>>>(Wv, Wqkv + 2 * HH, HH);
    k_cvt_f32_bf16<<<g4(HH), TPB, 0, stream>>>(Wo, Wob,           HH);
    (void)hipMemcpyAsync(bqkv,         bq, H * sizeof(float), hipMemcpyDeviceToDevice, stream);
    (void)hipMemcpyAsync(bqkv + H,     bk, H * sizeof(float), hipMemcpyDeviceToDevice, stream);
    (void)hipMemcpyAsync(bqkv + 2 * H, bv, H * sizeof(float), hipMemcpyDeviceToDevice, stream);
    k_avg3_bf16<<<(HH + TPB - 1) / TPB, TPB, 0, stream>>>(Ws, Wavg, HH, HH);
    k_avg3_f32 <<<(H  + TPB - 1) / TPB, TPB, 0, stream>>>(bs, bavg, H,  H);

    /* fused QKV projection: [4096,1024] x [3072,1024]^T */
    gemm_bias_wmma<0><<<dim3(3 * H / BN, MROWS / BM), 256, 0, stream>>>(
        Xbf, Wqkv, bqkv, QKV, MROWS, 3 * H, H);

    /* flash attention */
    flash_attn<<<dim3(SEQ / FB, NH, BATCH), 128, 0, stream>>>(QKV, Attn);

    /* strata-mean projection (folded to one affine map) */
    gemm_bias_wmma<0><<<dim3(H / BN, MROWS / BM), 256, 0, stream>>>(
        Attn, Wavg, bavg, Enh, MROWS, H, H);

    /* output projection -> fp32 d_out */
    gemm_bias_wmma<1><<<dim3(H / BN, MROWS / BM), 256, 0, stream>>>(
        Enh, Wob, bo, d_out, MROWS, H, H);
}
